// DPSNR_17600775979585
// MI455X (gfx1250) — matrix operand order repaired and checked by reading, not compiled, not run on hardware
//
#include <hip/hip_runtime.h>
#include <math.h>

// ---------------- model constants ----------------
static constexpr int Bc = 4;
static constexpr int Tc = 512;
static constexpr int Dc = 512;
static constexpr int NHc = 8;
static constexpr int HDc = 64;      // D / NH
static constexpr int FFc = 1024;
static constexpr int Vc = 50257;
static constexpr int NLc = 6;
static constexpr int POOLN = 65536;
static constexpr int WINc = 32;
static constexpr int LOOPS = 4;
static constexpr int ROWS = Bc * Tc;          // 2048
static constexpr float HALT_TH = 0.99f;

// ---------------- WMMA types ----------------
typedef __attribute__((ext_vector_type(16))) __bf16 bf16x16;
typedef __attribute__((ext_vector_type(8)))  float  f32x8;

union AFrag { bf16x16 v; unsigned int u[8]; };

__device__ __forceinline__ unsigned short f2bf(float f) {
  union { float f; unsigned int u; } c; c.f = f;
  unsigned int u = c.u;
  u += 0x7fffu + ((u >> 16) & 1u);   // round-to-nearest-even
  return (unsigned short)(u >> 16);
}
__device__ __forceinline__ unsigned int pack2bf(float a, float b) {
  return (unsigned int)f2bf(a) | ((unsigned int)f2bf(b) << 16);
}
__device__ __forceinline__ float geluf(float x) {
  float x3 = x * x * x;
  return 0.5f * x * (1.0f + tanhf(0.7978845608028654f * (x + 0.044715f * x3)));
}
__device__ __forceinline__ float sigmf(float x) { return 1.0f / (1.0f + __expf(-x)); }

// flags
#define GF_TRANSB 1
#define GF_GELU   2
#define GF_SIGM   4

// ---------------- generic WMMA GEMM ----------------
// C = act( scale * A@W + bias[n] + rowBias[row/rowDiv][n] + Res ) per flags.
// W is [K,N] row-major, or [N,K] row-major when TRANSB.
// Requirements honored by all call sites: K % 32 == 0; lda even; (TRANSB) ldb even.
// M/N ragged edges handled by address clamping (clamped lanes never stored).
// Batched over blockIdx.z: z -> (zo = z/zdiv, zi = z%zdiv).
#define BM 128
#define BK 32

template <bool TRANSB, int BNT>
__global__ __launch_bounds__(256)
void k_gemm(const float* __restrict__ A, const float* __restrict__ W,
            const float* __restrict__ bias, const float* __restrict__ rowBias,
            const float* __restrict__ Res, float* __restrict__ C,
            int M, int N, int K, int lda, int ldb, int ldc,
            long long oAo, long long oAi, long long oBo, long long oBi,
            long long oCo, long long oCi, int zdiv,
            int flags, float scale, int rowDiv)
{
  constexpr int NT = BNT / 32;   // 16-col tiles per wave (2 waves across N)

  int z = blockIdx.z;
  int zo = z / zdiv, zi = z % zdiv;
  A += (size_t)zo * oAo + (size_t)zi * oAi;
  W += (size_t)zo * oBo + (size_t)zi * oBi;
  size_t coff = (size_t)zo * oCo + (size_t)zi * oCi;
  C += coff;
  if (Res) Res += coff;

  __shared__ unsigned int As[2][BM][BK / 2 + 1];    // [buf][m][k2]
  __shared__ unsigned int Bs[2][BNT][BK / 2 + 1];   // [buf][n][k2]

  const int tid  = threadIdx.x;
  const int row0 = blockIdx.y * BM;
  const int col0 = blockIdx.x * BNT;
  const int wave = tid >> 5, lane = tid & 31;
  const int wM = wave >> 1, wN = wave & 1;          // 4x2 wave grid
  const int lm = lane & 15, lh = lane >> 4;

  // ---- unconditional, clamped staging (no divergence; OOB lanes never stored) ----
  auto stageA = [&](int b, int kk) {
#pragma unroll
    for (int i = 0; i < (BM * (BK / 2)) / 256; ++i) {
      int c  = tid + i * 256;
      int m  = c >> 4, k2 = c & 15;
      int gm = row0 + m; gm = (gm < M) ? gm : (M - 1);
      const float2 v = *(const float2*)(A + (size_t)gm * lda + kk + 2 * k2);
      As[b][m][k2] = pack2bf(v.x, v.y);
    }
  };
  auto stageB = [&](int b, int kk) {
#pragma unroll
    for (int i = 0; i < (BNT * (BK / 2)) / 256; ++i) {
      int c = tid + i * 256;
      if constexpr (TRANSB) {            // W[N][K]
        int n  = c >> 4, k2 = c & 15;
        int gn = col0 + n; gn = (gn < N) ? gn : (N - 1);
        const float2 v = *(const float2*)(W + (size_t)gn * ldb + kk + 2 * k2);
        Bs[b][n][k2] = pack2bf(v.x, v.y);
      } else {                           // W[K][N]
        int n  = c & (BNT - 1), k2 = c / BNT;
        int gn = col0 + n; gn = (gn < N) ? gn : (N - 1);
        const float* p = W + (size_t)(kk + 2 * k2) * ldb + gn;
        Bs[b][n][k2] = pack2bf(p[0], p[ldb]);
      }
    }
  };

  f32x8 acc[2][NT];
#pragma unroll
  for (int mt = 0; mt < 2; ++mt)
#pragma unroll
    for (int nt = 0; nt < NT; ++nt)
#pragma unroll
      for (int j = 0; j < 8; ++j) acc[mt][nt][j] = 0.0f;

  stageA(0, 0);
  stageB(0, 0);
  __syncthreads();

  int buf = 0;
  for (int k0 = 0; k0 < K; k0 += BK) {
    int nk = k0 + BK;
    if (nk < K) {
      stageA(buf ^ 1, nk);
      stageB(buf ^ 1, nk);
      if (nk + BK < K) {                 // prefetch tile after next
        int m = (tid >> 1) & (BM - 1);
        int gm = row0 + m; gm = (gm < M) ? gm : (M - 1);
        __builtin_prefetch(A + (size_t)gm * lda + nk + BK, 0, 1);
      }
    }

    // ---- fragments per ISA 16-bit VGPR layouts ----
    AFrag afr[2], bfr[NT];
#pragma unroll
    for (int mt = 0; mt < 2; ++mt) {
      int m = wM * 32 + mt * 16 + lm;
#pragma unroll
      for (int j = 0; j < 4; ++j) afr[mt].u[j]     = As[buf][m][4 * lh + j];
#pragma unroll
      for (int j = 0; j < 4; ++j) afr[mt].u[4 + j] = As[buf][m][8 + 4 * lh + j];
    }
#pragma unroll
    for (int nt = 0; nt < NT; ++nt) {
      int n = wN * (BNT / 2) + nt * 16 + lm;
#pragma unroll
      for (int j = 0; j < 8; ++j) bfr[nt].u[j] = Bs[buf][n][8 * (j >> 2) + 4 * lh + (j & 3)];
    }
#pragma unroll
    for (int mt = 0; mt < 2; ++mt)
#pragma unroll
      for (int nt = 0; nt < NT; ++nt)
        acc[mt][nt] = __builtin_amdgcn_wmma_f32_16x16x32_bf16(
            false, afr[mt].v, false, bfr[nt].v, (short)0, acc[mt][nt], false, false);

    __syncthreads();
    buf ^= 1;
  }

  // ---- epilogue ----
#pragma unroll
  for (int mt = 0; mt < 2; ++mt) {
#pragma unroll
    for (int nt = 0; nt < NT; ++nt) {
      int gn = col0 + wN * (BNT / 2) + nt * 16 + lm;
      if (gn >= N) continue;
      float bv = bias ? bias[gn] : 0.0f;
#pragma unroll
      for (int j = 0; j < 8; ++j) {
        int gm = row0 + wM * 32 + mt * 16 + j + 8 * lh;  // C: lanes0-15->M=j, 16-31->M=j+8
        if (gm >= M) continue;
        float v = acc[mt][nt][j] * scale + bv;
        if (rowBias) v += rowBias[(size_t)(gm / rowDiv) * N + gn];
        size_t o = (size_t)gm * ldc + gn;
        if (Res) v += Res[o];
        if (flags & GF_GELU) v = geluf(v);
        else if (flags & GF_SIGM) v = sigmf(v);
        C[o] = v;
      }
    }
  }
}

// ---------------- elementwise / reduction kernels ----------------
__global__ void k_embed(const int* __restrict__ ids, const float* __restrict__ emb,
                        const float* __restrict__ pos, float* __restrict__ x)
{
  int row = blockIdx.x;             // b*T + t
  int t   = row % Tc;
  int id  = ids[row];
  for (int d = threadIdx.x; d < Dc; d += blockDim.x)
    x[(size_t)row * Dc + d] = emb[(size_t)id * Dc + d] + pos[(size_t)t * Dc + d];
}

__global__ void k_ln(const float* __restrict__ in, const float* __restrict__ sc,
                     const float* __restrict__ bi, float* __restrict__ out, int dim)
{
  __shared__ float s1[256], s2[256];
  int row = blockIdx.x, tid = threadIdx.x;
  const float* r = in + (size_t)row * dim;
  float a = 0.0f, b = 0.0f;
  for (int d = tid; d < dim; d += 256) { float v = r[d]; a += v; b += v * v; }
  s1[tid] = a; s2[tid] = b; __syncthreads();
  for (int off = 128; off > 0; off >>= 1) {
    if (tid < off) { s1[tid] += s1[tid + off]; s2[tid] += s2[tid + off]; }
    __syncthreads();
  }
  float mean = s1[0] / dim;
  float var  = s2[0] / dim - mean * mean;
  float inv  = rsqrtf(var + 1e-6f);
  float* w = out + (size_t)row * dim;
  for (int d = tid; d < dim; d += 256)
    w[d] = (r[d] - mean) * inv * sc[d] + bi[d];
}

__global__ void k_softmax_causal(float* __restrict__ sc)
{
  __shared__ float red[256];
  int t = blockIdx.x, z = blockIdx.y, tid = threadIdx.x;
  float* row = sc + ((size_t)z * Tc + t) * Tc;
  int lim = t + 1;
  float mx = -1e30f;
  for (int j = tid; j < lim; j += 256) mx = fmaxf(mx, row[j]);
  red[tid] = mx; __syncthreads();
  for (int off = 128; off > 0; off >>= 1) {
    if (tid < off) red[tid] = fmaxf(red[tid], red[tid + off]);
    __syncthreads();
  }
  mx = red[0]; __syncthreads();
  float sum = 0.0f;
  for (int j = tid; j < lim; j += 256) { float e = __expf(row[j] - mx); row[j] = e; sum += e; }
  red[tid] = sum; __syncthreads();
  for (int off = 128; off > 0; off >>= 1) {
    if (tid < off) red[tid] += red[tid + off];
    __syncthreads();
  }
  float inv = 1.0f / red[0];
  for (int j = tid; j < Tc; j += 256) row[j] = (j < lim) ? row[j] * inv : 0.0f;
}

// out[b,n] = act( A[b]@W + bias[n] ); A rows at A + b*lda, W is [K,N] row-major.
__global__ void k_small_dense(const float* __restrict__ A, int lda,
                              const float* __restrict__ W, const float* __restrict__ bias,
                              float* __restrict__ out, int Bn, int N, int K, int act)
{
  int i = blockIdx.x * blockDim.x + threadIdx.x;
  if (i >= Bn * N) return;
  int b = i / N, n = i % N;
  const float* a = A + (size_t)b * lda;
  float acc = bias ? bias[n] : 0.0f;
  for (int k = 0; k < K; ++k) acc += a[k] * W[(size_t)k * N + n];
  if (act == 1) acc = geluf(acc);
  else if (act == 2) acc = sigmf(acc);
  else if (act == 3) acc = log1pf(__expf(acc));  // softplus
  out[i] = acc;
}

__global__ void k_pool(const float* __restrict__ mu, const float* __restrict__ sg,
                       const float* __restrict__ pool, float* __restrict__ out)
{
  __shared__ float w[WINc];
  __shared__ float sinv;
  int b = blockIdx.x, tid = threadIdx.x;
  float center = mu[b] * (float)(POOLN - 1);
  float cl = fminf(fmaxf(center - (float)(WINc / 2), 0.0f), (float)(POOLN - WINc));
  int start = (int)cl;
  if (tid < WINc) {
    float dist = (float)(start + tid) - center;
    float s = sg[b] + 1e-6f;
    w[tid] = __expf(-dist * dist / (2.0f * s * s));
  }
  __syncthreads();
  if (tid == 0) {
    float s = 0.0f;
    for (int j = 0; j < WINc; ++j) s += w[j];
    sinv = 1.0f / (s + 1e-6f);
  }
  __syncthreads();
  for (int d = tid; d < Dc; d += blockDim.x) {
    float acc = 0.0f;
    for (int j = 0; j < WINc; ++j)
      acc += w[j] * pool[(size_t)(start + j) * Dc + d];
    out[(size_t)b * Dc + d] = acc * sinv;
  }
}

__global__ void k_stepout(const float* __restrict__ s, const float* __restrict__ integ,
                          const float* __restrict__ loop_emb, float* __restrict__ out, int iter)
{
  size_t i = (size_t)blockIdx.x * blockDim.x + threadIdx.x;
  if (i >= (size_t)ROWS * Dc) return;
  out[i] = s[i] + integ[i] + loop_emb[(size_t)iter * Dc + (i & (Dc - 1))];
}

__global__ void k_gate(const float* __restrict__ g, const float* __restrict__ t,
                       const float* __restrict__ s, float* __restrict__ out)
{
  size_t i = (size_t)blockIdx.x * blockDim.x + threadIdx.x;
  if (i >= (size_t)ROWS * Dc) return;
  float gv = g[i];
  out[i] = gv * t[i] + (1.0f - gv) * s[i];
}

__global__ void k_halt(const float* __restrict__ hh, const float* __restrict__ W,
                       const float* __restrict__ bias, float* __restrict__ halt)
{
  __shared__ float sh[128];
  int r = blockIdx.x, tid = threadIdx.x;
  sh[tid] = hh[(size_t)r * 128 + tid] * W[tid];
  __syncthreads();
  for (int off = 64; off > 0; off >>= 1) {
    if (tid < off) sh[tid] += sh[tid + off];
    __syncthreads();
  }
  if (tid == 0) halt[r] = sigmf(sh[0] + bias[0]);
}

__global__ void k_supd(float* __restrict__ s, const float* __restrict__ cand,
                       const float* __restrict__ hm)
{
  size_t i = (size_t)blockIdx.x * blockDim.x + threadIdx.x;
  if (i >= (size_t)ROWS * Dc) return;
  float m = hm[i / Dc];
  s[i] = (1.0f - m) * cand[i] + m * s[i];
}

__global__ void k_hupd(float* __restrict__ hp, float* __restrict__ hm,
                       const float* __restrict__ halt)
{
  int r = blockIdx.x * blockDim.x + threadIdx.x;
  if (r >= ROWS) return;
  float m = hm[r];
  float nhp = hp[r] + halt[r] * (1.0f - m);
  hp[r] = nhp;
  hm[r] = fmaxf(m, (nhp >= HALT_TH) ? 1.0f : 0.0f);
}

__global__ void k_zero(float* __restrict__ p, int n)
{
  int i = blockIdx.x * blockDim.x + threadIdx.x;
  if (i < n) p[i] = 0.0f;
}

// ---------------- host-side orchestration ----------------
static inline int cdiv(int a, int b) { return (a + b - 1) / b; }

static void gemm(hipStream_t st, const float* A, const float* W, const float* bias,
                 const float* rowBias, const float* Res, float* C,
                 int M, int N, int K, int lda, int ldb, int ldc,
                 int batches, int zdiv,
                 long long oAo, long long oAi, long long oBo, long long oBi,
                 long long oCo, long long oCi,
                 int flags, float scale, int rowDiv)
{
  bool trans = (flags & GF_TRANSB) != 0;
  int bnt = (trans || N >= 128) ? 128 : 64;
  dim3 g(cdiv(N, bnt), cdiv(M, BM), batches);
  if (trans)
    k_gemm<true, 128><<<g, 256, 0, st>>>(A, W, bias, rowBias, Res, C, M, N, K, lda, ldb, ldc,
                                         oAo, oAi, oBo, oBi, oCo, oCi, zdiv, flags, scale, rowDiv);
  else if (bnt == 128)
    k_gemm<false, 128><<<g, 256, 0, st>>>(A, W, bias, rowBias, Res, C, M, N, K, lda, ldb, ldc,
                                          oAo, oAi, oBo, oBi, oCo, oCi, zdiv, flags, scale, rowDiv);
  else
    k_gemm<false, 64><<<g, 256, 0, st>>>(A, W, bias, rowBias, Res, C, M, N, K, lda, ldb, ldc,
                                         oAo, oAi, oBo, oBi, oCo, oCi, zdiv, flags, scale, rowDiv);
}

// params (pytree / alphabetical flatten, input_ids first)
enum {
  I_IDS = 0, I_WLM, I_AWG, I_AWH1, I_AWH2, I_AWT, I_ABG, I_ABH1, I_ABH2, I_ABT,
  I_ALNB, I_ALNS, I_ALOOP, I_EMB, I_FNB, I_FNS,
  I_XW1, I_XW2, I_XWMU, I_XWSG, I_XB1, I_XB2, I_XBMU, I_XBSG,
  I_LAYER0,                 // 24: 6 layers x {W1,W2,Wo,Wqkv,b1,b2,ln1_b,ln1_s,ln2_b,ln2_s}
  I_POOL = I_LAYER0 + NLc * 10,  // 84
  I_POS, I_RIW1, I_RIW2, I_RIB1, I_RIB2, I_RILNB, I_RILNS
};

extern "C" void kernel_launch(void* const* d_in, const int* in_sizes, int n_in,
                              void* d_out, int out_size, void* d_ws, size_t ws_size,
                              hipStream_t stream)
{
  (void)in_sizes; (void)n_in; (void)out_size; (void)ws_size;
  auto P = [&](int i) { return (const float*)d_in[i]; };
  const int* ids = (const int*)d_in[I_IDS];

  float* ws = (float*)d_ws;
  const size_t U = 1048576;                  // 1M floats
  float* x      = ws;                        // [2048,512]  (doubles as loop state s)
  float* h      = ws + 1 * U;                // LN scratch
  float* qkv    = ws + 2 * U;                // [2048,1536]
  float* scores = ws + 5 * U;                // [32,512,512] (8M)
  float* attn   = ws + 13 * U;               // [2048,512]
  float* ffh    = ws + 14 * U;               // [2048,1024]
  // loop-phase aliases over dead encoder scratch
  float* bufA = qkv;                         // integ (gelu out)
  float* bufB = qkv + U;                     // step_out
  float* bufC = qkv + 2 * U;                 // gate g
  float* bufD = scores;                      // gemm temp
  float* bufE = scores + U;                  // cand
  float* bufF = scores + 2 * U;              // LN outputs (integ_ln / cand_ln / final)
  float* hh   = scores + 3 * U;              // [2048,128]
  float* halt = hh + ROWS * 128;
  float* hp   = halt + ROWS;
  float* hm   = hp + ROWS;
  float* muv  = hm + ROWS;
  float* sgv  = muv + Bc;
  float* retr = sgv + Bc;                    // [4,512]
  float* r2   = retr + Bc * Dc;              // [4,512]
  float* x1   = r2 + Bc * Dc;                // [4,512]
  float* x2   = x1 + Bc * Dc;                // [4,256]

  // ---------- encoder ----------
  k_embed<<<ROWS, 256, 0, stream>>>(ids, P(I_EMB), P(I_POS), x);

  for (int l = 0; l < NLc; ++l) {
    const float* W1   = P(I_LAYER0 + l * 10 + 0);
    const float* W2   = P(I_LAYER0 + l * 10 + 1);
    const float* Wo   = P(I_LAYER0 + l * 10 + 2);
    const float* Wqkv = P(I_LAYER0 + l * 10 + 3);
    const float* b1   = P(I_LAYER0 + l * 10 + 4);
    const float* b2   = P(I_LAYER0 + l * 10 + 5);
    const float* l1b  = P(I_LAYER0 + l * 10 + 6);
    const float* l1s  = P(I_LAYER0 + l * 10 + 7);
    const float* l2b  = P(I_LAYER0 + l * 10 + 8);
    const float* l2s  = P(I_LAYER0 + l * 10 + 9);

    k_ln<<<ROWS, 256, 0, stream>>>(x, l1s, l1b, h, Dc);
    // qkv = h @ Wqkv
    gemm(stream, h, Wqkv, nullptr, nullptr, nullptr, qkv,
         ROWS, 3 * Dc, Dc, Dc, 3 * Dc, 3 * Dc, 1, 1, 0, 0, 0, 0, 0, 0, 0, 1.0f, 1);
    // scores[z] = q @ k^T / 8   (z = b*NH + h)
    gemm(stream, qkv, qkv + Dc, nullptr, nullptr, nullptr, scores,
         Tc, Tc, HDc, 3 * Dc, 3 * Dc, Tc, Bc * NHc, NHc,
         (long long)Tc * 3 * Dc, HDc, (long long)Tc * 3 * Dc, HDc,
         (long long)NHc * Tc * Tc, (long long)Tc * Tc,
         GF_TRANSB, 0.125f, 1);
    dim3 smg(Tc, Bc * NHc);
    k_softmax_causal<<<smg, 256, 0, stream>>>(scores);
    // attn[b,:,h*HD:] = probs @ v
    gemm(stream, scores, qkv + 2 * Dc, nullptr, nullptr, nullptr, attn,
         Tc, HDc, Tc, Tc, 3 * Dc, Dc, Bc * NHc, NHc,
         (long long)NHc * Tc * Tc, (long long)Tc * Tc,
         (long long)Tc * 3 * Dc, HDc,
         (long long)Tc * Dc, HDc,
         0, 1.0f, 1);
    // x = x + attn @ Wo
    gemm(stream, attn, Wo, nullptr, nullptr, x, x,
         ROWS, Dc, Dc, Dc, Dc, Dc, 1, 1, 0, 0, 0, 0, 0, 0, 0, 1.0f, 1);
    // FFN
    k_ln<<<ROWS, 256, 0, stream>>>(x, l2s, l2b, h, Dc);
    gemm(stream, h, W1, b1, nullptr, nullptr, ffh,
         ROWS, FFc, Dc, Dc, FFc, FFc, 1, 1, 0, 0, 0, 0, 0, 0, GF_GELU, 1.0f, 1);
    gemm(stream, ffh, W2, b2, nullptr, x, x,
         ROWS, Dc, FFc, FFc, Dc, Dc, 1, 1, 0, 0, 0, 0, 0, 0, 0, 1.0f, 1);
  }

  // ---------- adaptive compute loop ----------
  k_zero<<<cdiv(2 * ROWS, 256), 256, 0, stream>>>(hp, 2 * ROWS);  // hp and hm contiguous

  for (int it = 0; it < LOOPS; ++it) {
    // indexer on last-token state s[:, T-1, :]
    const float* slast = x + (size_t)(Tc - 1) * Dc;   // + b*(T*D) via lda
    k_small_dense<<<cdiv(Bc * Dc, 256), 256, 0, stream>>>(
        slast, Tc * Dc, P(I_XW1), P(I_XB1), x1, Bc, Dc, Dc, 1);
    k_small_dense<<<cdiv(Bc * (Dc / 2), 256), 256, 0, stream>>>(
        x1, Dc, P(I_XW2), P(I_XB2), x2, Bc, Dc / 2, Dc, 1);
    k_small_dense<<<1, 64, 0, stream>>>(x2, Dc / 2, P(I_XWMU), P(I_XBMU), muv, Bc, 1, Dc / 2, 2);
    k_small_dense<<<1, 64, 0, stream>>>(x2, Dc / 2, P(I_XWSG), P(I_XBSG), sgv, Bc, 1, Dc / 2, 3);
    // pooled retrieval (gather over 65536x512 table)
    k_pool<<<Bc, 256, 0, stream>>>(muv, sgv, P(I_POOL), retr);
    // r2[b] = retr[b] @ ri_W1[D:,:] + ri_b1
    k_small_dense<<<cdiv(Bc * Dc, 256), 256, 0, stream>>>(
        retr, Dc, P(I_RIW1) + (size_t)Dc * Dc, P(I_RIB1), r2, Bc, Dc, Dc, 0);
    // integ = gelu(s @ ri_W1[:D] + r2[row-batch])
    gemm(stream, x, P(I_RIW1), nullptr, r2, nullptr, bufA,
         ROWS, Dc, Dc, Dc, Dc, Dc, 1, 1, 0, 0, 0, 0, 0, 0, GF_GELU, 1.0f, Tc);
    // integ2 = integ @ ri_W2 + ri_b2 ; integ_ln = LN(integ2)
    gemm(stream, bufA, P(I_RIW2), P(I_RIB2), nullptr, nullptr, bufD,
         ROWS, Dc, Dc, Dc, Dc, Dc, 1, 1, 0, 0, 0, 0, 0, 0, 0, 1.0f, 1);
    k_ln<<<ROWS, 256, 0, stream>>>(bufD, P(I_RILNS), P(I_RILNB), bufF, Dc);
    // step_out = s + integ_ln + loop_emb[it]
    k_stepout<<<cdiv(ROWS * Dc, 256), 256, 0, stream>>>(x, bufF, P(I_ALOOP), bufB, it);
    // g = sigmoid(step_out @ Wg[:D] + s @ Wg[D:] + bg)
    gemm(stream, bufB, P(I_AWG), nullptr, nullptr, nullptr, bufC,
         ROWS, Dc, Dc, Dc, Dc, Dc, 1, 1, 0, 0, 0, 0, 0, 0, 0, 1.0f, 1);
    gemm(stream, x, P(I_AWG) + (size_t)Dc * Dc, P(I_ABG), nullptr, bufC, bufC,
         ROWS, Dc, Dc, Dc, Dc, Dc, 1, 1, 0, 0, 0, 0, 0, 0, GF_SIGM, 1.0f, 1);
    // t = step_out @ Wt + bt
    gemm(stream, bufB, P(I_AWT), P(I_ABT), nullptr, nullptr, bufD,
         ROWS, Dc, Dc, Dc, Dc, Dc, 1, 1, 0, 0, 0, 0, 0, 0, 0, 1.0f, 1);
    // cand = g*t + (1-g)*s ; cand_ln = LN(cand)
    k_gate<<<cdiv(ROWS * Dc, 256), 256, 0, stream>>>(bufC, bufD, x, bufE);
    k_ln<<<ROWS, 256, 0, stream>>>(bufE, P(I_ALNS), P(I_ALNB), bufF, Dc);
    // halt = sigmoid(gelu(cand_ln @ Wh1 + bh1) @ Wh2 + bh2)
    gemm(stream, bufF, P(I_AWH1), P(I_ABH1), nullptr, nullptr, hh,
         ROWS, Dc / 4, Dc, Dc, Dc / 4, Dc / 4, 1, 1, 0, 0, 0, 0, 0, 0, GF_GELU, 1.0f, 1);
    k_halt<<<ROWS, 128, 0, stream>>>(hh, P(I_AWH2), P(I_ABH2), halt);
    // s = (1-hm_old)*cand_ln + hm_old*s ; then update hp/hm
    k_supd<<<cdiv(ROWS * Dc, 256), 256, 0, stream>>>(x, bufF, hm);
    k_hupd<<<cdiv(ROWS, 256), 256, 0, stream>>>(hp, hm, halt);
  }

  // ---------- LM head ----------
  k_ln<<<ROWS, 256, 0, stream>>>(x, P(I_FNS), P(I_FNB), bufF, Dc);
  gemm(stream, bufF, P(I_WLM), nullptr, nullptr, nullptr, (float*)d_out,
       ROWS, Vc, Dc, Dc, Vc, Vc, 1, 1, 0, 0, 0, 0, 0, 0, 0, 1.0f, 1);
}
